// LocalSelfAttention_15255723835982
// MI455X (gfx1250) — matrix operand order repaired and checked by reading, not compiled, not run on hardware
//
#include <hip/hip_runtime.h>

// ---------------------------------------------------------------------------
// LocalSelfAttention (Swin-window attention) for MI455X / gfx1250, wave32.
//   K0: weights f32 -> bf16
//   K1: per-window [pack/scramble -> LDS] + QKV GEMM (wmma bf16) -> qkv global
//   K2: per-window [16-head attention (V via TDM -> LDS, LDS softmax)
//                   -> o in LDS -> proj GEMM + bias + window-merge scatter]
// HBM traffic ~652MB (x + qkv round-trip + out) ~= 28us at 23.3 TB/s.
// ---------------------------------------------------------------------------

typedef __bf16 bf16;
typedef float        f32x4  __attribute__((ext_vector_type(4)));
typedef unsigned int u32x4  __attribute__((ext_vector_type(4)));
typedef int          i32x4  __attribute__((ext_vector_type(4)));
typedef int          i32x8  __attribute__((ext_vector_type(8)));
typedef bf16         bf16x4 __attribute__((ext_vector_type(4)));
typedef bf16         v16bf  __attribute__((ext_vector_type(16)));
typedef float        v8f    __attribute__((ext_vector_type(8)));

#define WMMA_BF16(a, b, c) \
  __builtin_amdgcn_wmma_f32_16x16x32_bf16(false, (a), false, (b), (short)0, (c), false, false)

#define NWIN    1024
#define TOK     64
#define C_DIM   512
#define NH      16
#define HD      32
#define C3      1536
#define SCALE   0.17677669529663687f   // 1/sqrt(32)

// Workspace (bytes): wq | wp | qkv   (~203 MB total)
#define OFF_WQKV  ((size_t)0)           // 1536*512  bf16
#define OFF_WPROJ ((size_t)1572864)     // 512*512   bf16
#define OFF_QKV   ((size_t)2097152)     // 65536*1536 bf16 = 192 MB

#if defined(__has_builtin)
#if __has_builtin(__builtin_amdgcn_tensor_load_to_lds) && \
    __has_builtin(__builtin_amdgcn_s_wait_tensorcnt)
#define HAVE_TDM 1
#endif
#endif

union FragU { v16bf v; u32x4 u[2]; bf16 h[16]; };

// A fragment: 16x32 bf16, row-major source (M rows contiguous over K).
// lanes0-15=M, contiguous K order: element s of lane half h holds K=16h+s.
static __device__ __forceinline__ v16bf load_a_frag(const bf16* A, int ld, int m_base,
                                                    int k0, int lane) {
  const int half = lane >> 4;
  const bf16* row = A + (size_t)(m_base + (lane & 15)) * (size_t)ld;
  FragU f;
  f.u[0] = *(const u32x4*)(row + k0 + (half ? 16 : 0));
  f.u[1] = *(const u32x4*)(row + k0 + 8 + (half ? 16 : 0));
  return f.v;
}

// B fragment: 32x16 bf16 from N-major storage Bn[n][k] (K contiguous).
static __device__ __forceinline__ v16bf load_b_frag(const bf16* Bn, int ld, int n_base,
                                                    int k0, int lane) {
  const int half = lane >> 4;
  const bf16* row = Bn + (size_t)(n_base + (lane & 15)) * (size_t)ld;
  const int kb = k0 + (half ? 16 : 0);
  FragU f;
  f.u[0] = *(const u32x4*)(row + kb);
  f.u[1] = *(const u32x4*)(row + kb + 8);
  return f.v;
}

// B fragment from K-major storage Bk[k][n] (V in natural (token, hd) layout).
static __device__ __forceinline__ v16bf load_b_frag_kmajor(const bf16* Bk, int ld,
                                                           int n_base, int k0, int lane) {
  const int half = lane >> 4;
  const int n = n_base + (lane & 15);
  const int kb = k0 + (half ? 16 : 0);
  FragU f;
#pragma unroll
  for (int i = 0; i < 8; ++i) {
    f.h[2 * i]     = Bk[(size_t)(kb + 2 * i) * ld + n];
    f.h[2 * i + 1] = Bk[(size_t)(kb + 2 * i + 1) * ld + n];
  }
  return f.v;
}

// ---------------------------------------------------------------------------
// K0: weights f32 -> bf16 (natural [out_ch][in_ch] = N-major layout)
// ---------------------------------------------------------------------------
__global__ __launch_bounds__(256) void lsa_pack_w(const float* __restrict__ w_qkv,
                                                  const float* __restrict__ w_proj,
                                                  bf16* __restrict__ wq,
                                                  bf16* __restrict__ wp) {
  const int i = blockIdx.x * 256 + threadIdx.x;
  const int NQ = C3 * C_DIM, NP = C_DIM * C_DIM;
  if (i < NQ)           wq[i] = (bf16)w_qkv[i];
  else if (i < NQ + NP) wp[i - NQ] = (bf16)w_proj[i - NQ];
}

// ---------------------------------------------------------------------------
// K1: per-window pack (unfold+view scramble, f32->bf16, direct into LDS) then
// QKV GEMM: qkv[64 x 1536] = xw_lds[64 x 512] * wq^T.  256 thr = 8 waves:
// waves split 4(M-slices of 16) x 2(N-halves of 768); 12 n-groups of 64 each.
// ---------------------------------------------------------------------------
__global__ __launch_bounds__(256) void lsa_win_qkv(const float* __restrict__ x,
                                                   const bf16* __restrict__ wq,
                                                   bf16* __restrict__ qkv) {
  __shared__ __align__(16) bf16 xw[64][520];   // scrambled window tile, padded
  const int win = blockIdx.x;
  const int b = win >> 6, wh = (win >> 3) & 7, ww = win & 7;
  const int tid = threadIdx.x;

  // pack: coalesced f32x4 reads of the window slab; scrambled bf16 LDS writes.
  // xw[p={t,s}][c'={c&7,i,j}] = x[b, (wh*8+i)*64 + ww*8+j, c]
  for (int idx = tid; idx < 64 * 128; idx += 256) {
    const int spatial = idx >> 7;              // i*8 + j
    const int c4 = (idx & 127) * 4;
    const int i = spatial >> 3, j = spatial & 7;
    const size_t src = ((size_t)(b * 4096 + (wh * 8 + i) * 64 + ww * 8 + j)) * C_DIM + c4;
    const f32x4 s = *(const f32x4*)(x + src);
    const int p = ((c4 >> 3) & 7) * 8 + (c4 >> 6);     // t*8 + s
    xw[p][(((c4 + 0) & 7) << 6) | spatial] = (bf16)s.x;
    xw[p][(((c4 + 1) & 7) << 6) | spatial] = (bf16)s.y;
    xw[p][(((c4 + 2) & 7) << 6) | spatial] = (bf16)s.z;
    xw[p][(((c4 + 3) & 7) << 6) | spatial] = (bf16)s.w;
  }
  __syncthreads();

  const int lane = tid & 31, wave = tid >> 5;
  const int m0 = (wave & 3) * 16;
  const int nbase = (wave >> 2) * 768;
  const int half = lane >> 4, nl = lane & 15;
  bf16* Dw = qkv + (size_t)win * 64 * C3;

  for (int g = 0; g < 12; ++g) {
    const int n0 = nbase + g * 64;
    v8f acc[4] = {{0,0,0,0,0,0,0,0},{0,0,0,0,0,0,0,0},
                  {0,0,0,0,0,0,0,0},{0,0,0,0,0,0,0,0}};
#pragma unroll 4
    for (int k0 = 0; k0 < C_DIM; k0 += 32) {
      const v16bf a = load_a_frag(&xw[0][0], 520, m0, k0, lane);
#pragma unroll
      for (int nt = 0; nt < 4; ++nt) {
        const v16bf bfr = load_b_frag(wq, C_DIM, n0 + nt * 16, k0, lane);
        acc[nt] = WMMA_BF16(a, bfr, acc[nt]);
      }
    }
#pragma unroll
    for (int nt = 0; nt < 4; ++nt)
#pragma unroll
      for (int v = 0; v < 8; ++v)
        Dw[(size_t)(m0 + v + half * 8) * C3 + n0 + nt * 16 + nl] = (bf16)acc[nt][v];
  }
}

// ---------------------------------------------------------------------------
// K2: per-window attention (16 heads serial) + output proj + merge scatter.
// 128 thr = 4 waves; wave w owns rows 16w..16w+15 in the WMMA phases.
// ---------------------------------------------------------------------------
__global__ __launch_bounds__(128) void lsa_attn_proj(const bf16* __restrict__ qkv,
                                                     const bf16* __restrict__ wp,
                                                     const float* __restrict__ bias,
                                                     float* __restrict__ out) {
  __shared__ __align__(16) bf16  v_lds[64][32];    //  4 KB  V tile (TDM target)
  __shared__ __align__(16) float s_lds[64][68];    // 17.4KB scores f32
  __shared__ __align__(16) bf16  p_lds[64][72];    //  9.2KB probs bf16
  __shared__ __align__(16) bf16  o_lds[64][520];   // 66.6KB attention output

  const int win = blockIdx.x;
  const int tid = threadIdx.x, lane = tid & 31, wave = tid >> 5;
  const int half = lane >> 4, nl = lane & 15;
  const bf16* base = qkv + (size_t)win * 64 * C3;

  for (int h = 0; h < NH; ++h) {
    const bf16* Q  = base + h * HD;
    const bf16* Kp = Q + C_DIM;
    const bf16* Vp = Q + 2 * C_DIM;

    // ---- stage V (64 x 32 bf16, row stride C3) into LDS ----
#ifdef HAVE_TDM
    if (tid < 32) {                      // wave 0 drives the Tensor Data Mover
      const unsigned long long ga = (unsigned long long)(size_t)Vp;
      const unsigned int la = (unsigned int)(size_t)&v_lds[0][0];
      u32x4 g0;
      g0.x = 1u;                                   // count=1, user D#
      g0.y = la;                                   // lds_addr
      g0.z = (unsigned int)ga;                     // global_addr[31:0]
      g0.w = (unsigned int)((ga >> 32) & 0x01FFFFFFu) | 0x80000000u; // addr hi | type=2
      i32x8 g1;
      g1.s0 = 1 << 16;                             // data_size=2B, no mask/pad
      g1.s1 = (int)((C3 & 0xFFFF) << 16);          // tensor_dim0 = 1536 (low16)
      g1.s2 = (int)((TOK * NWIN & 0xFFFF) << 16);  // dim0 hi=0 | tensor_dim1 lo16
      g1.s3 = (int)((TOK * NWIN) >> 16) | (HD << 16); // dim1 hi | tile_dim0=32
      g1.s4 = TOK;                                 // tile_dim1=64, tile_dim2=0
      g1.s5 = C3;                                  // tensor_dim0_stride = 1536
      g1.s6 = 0;                                   // stride hi | dim1_stride lo
      g1.s7 = 0;
      const i32x4 z4 = {0, 0, 0, 0};
#if __clang_major__ >= 23
      const i32x8 z8 = {0, 0, 0, 0, 0, 0, 0, 0};
      __builtin_amdgcn_tensor_load_to_lds(g0, g1, z4, z4, z8, 0);
#else
      __builtin_amdgcn_tensor_load_to_lds(g0, g1, z4, z4, 0);
#endif
      __builtin_amdgcn_s_wait_tensorcnt(0);
    }
#else
    {
      const int r = tid >> 1, seg = (tid & 1) * 16;
      const u32x4* src = (const u32x4*)(Vp + (size_t)r * C3 + seg);
      *(u32x4*)&v_lds[r][seg]     = src[0];
      *(u32x4*)&v_lds[r][seg + 8] = src[1];
    }
#endif
    __syncthreads();

    // ---- S = q k^T * scale (A=q rows, B=k rows N-major over hd) ----
    const v16bf aq = load_a_frag(Q, C3, wave * 16, 0, lane);
#pragma unroll
    for (int nt = 0; nt < 4; ++nt) {
      const v16bf bk = load_b_frag(Kp, C3, nt * 16, 0, lane);
      v8f s = {0,0,0,0,0,0,0,0};
      s = WMMA_BF16(aq, bk, s);
#pragma unroll
      for (int v = 0; v < 8; ++v)
        s_lds[wave * 16 + v + half * 8][nt * 16 + nl] = s[v] * SCALE;
    }
    __syncthreads();

    // ---- softmax, one thread per row; exp stored back to LDS (no spills) ----
    if (tid < 64) {
      float mx = -3.4e38f;
#pragma unroll 8
      for (int j = 0; j < 64; ++j) mx = fmaxf(mx, s_lds[tid][j]);
      float sum = 0.f;
#pragma unroll 8
      for (int j = 0; j < 64; ++j) {
        const float e = __expf(s_lds[tid][j] - mx);
        s_lds[tid][j] = e;
        sum += e;
      }
      const float inv = 1.f / sum;
#pragma unroll 8
      for (int j = 0; j < 64; ++j) p_lds[tid][j] = (bf16)(s_lds[tid][j] * inv);
    }
    __syncthreads();

    // ---- O = P V (K-dim = 64 tokens, two 32-chunks) -> o_lds[., h*32+.] ----
    v8f oacc[2] = {{0,0,0,0,0,0,0,0},{0,0,0,0,0,0,0,0}};
#pragma unroll
    for (int kt = 0; kt < 2; ++kt) {
      const v16bf ap = load_a_frag(&p_lds[0][0], 72, wave * 16, kt * 32, lane);
#pragma unroll
      for (int nt = 0; nt < 2; ++nt) {
        const v16bf bv = load_b_frag_kmajor(&v_lds[0][0], 32, nt * 16, kt * 32, lane);
        oacc[nt] = WMMA_BF16(ap, bv, oacc[nt]);
      }
    }
#pragma unroll
    for (int nt = 0; nt < 2; ++nt)
#pragma unroll
      for (int v = 0; v < 8; ++v)
        o_lds[wave * 16 + v + half * 8][h * HD + nt * 16 + nl] = (bf16)oacc[nt][v];
    __syncthreads();   // protects v_lds/s_lds/p_lds reuse and o_lds visibility
  }

  // ---- proj: out = o_lds * wp^T + bias, fused window-merge scatter ----
  const int b = win >> 6, wh = (win >> 3) & 7, ww = win & 7;
  const int m0 = wave * 16;
  for (int g = 0; g < 8; ++g) {
    const int n0 = g * 64;
    v8f acc[4] = {{0,0,0,0,0,0,0,0},{0,0,0,0,0,0,0,0},
                  {0,0,0,0,0,0,0,0},{0,0,0,0,0,0,0,0}};
#pragma unroll 4
    for (int k0 = 0; k0 < C_DIM; k0 += 32) {
      const v16bf a = load_a_frag(&o_lds[0][0], 520, m0, k0, lane);
#pragma unroll
      for (int nt = 0; nt < 4; ++nt) {
        const v16bf bfr = load_b_frag(wp, C_DIM, n0 + nt * 16, k0, lane);
        acc[nt] = WMMA_BF16(a, bfr, acc[nt]);
      }
    }
#pragma unroll
    for (int nt = 0; nt < 4; ++nt) {
      const int c = n0 + nt * 16 + nl;
      const float bv = bias[c];
#pragma unroll
      for (int v = 0; v < 8; ++v) {
        const int p = m0 + v + half * 8;
        const int hh = wh * 8 + (p >> 3), wcol = ww * 8 + (p & 7);
        out[((size_t)b * 4096 + hh * 64 + wcol) * C_DIM + c] = acc[nt][v] + bv;
      }
    }
  }
}

// ---------------------------------------------------------------------------
extern "C" void kernel_launch(void* const* d_in, const int* in_sizes, int n_in,
                              void* d_out, int out_size, void* d_ws, size_t ws_size,
                              hipStream_t stream) {
  const float* x      = (const float*)d_in[0];
  const float* w_qkv  = (const float*)d_in[1];
  const float* w_proj = (const float*)d_in[2];
  const float* b_proj = (const float*)d_in[3];
  float* out = (float*)d_out;

  char* wsb = (char*)d_ws;
  bf16* wq   = (bf16*)(wsb + OFF_WQKV);
  bf16* wp   = (bf16*)(wsb + OFF_WPROJ);
  bf16* qkvb = (bf16*)(wsb + OFF_QKV);

  (void)in_sizes; (void)n_in; (void)out_size; (void)ws_size;

  lsa_pack_w<<<(C3 * C_DIM + C_DIM * C_DIM + 255) / 256, 256, 0, stream>>>(w_qkv, w_proj, wq, wp);
  lsa_win_qkv<<<NWIN, 256, 0, stream>>>(x, wq, qkvb);
  lsa_attn_proj<<<NWIN, 128, 0, stream>>>(qkvb, wp, b_proj, out);
}
